// DeepBilateralNetCurves_34136400068722
// MI455X (gfx1250) — compile-verified, matrix operand order checked by reading, not run on hardware
//
#include <hip/hip_runtime.h>
#include <hip/hip_bf16.h>

typedef __attribute__((ext_vector_type(16))) _Float16 v16h;
typedef __attribute__((ext_vector_type(8)))  float    v8f;

// CDNA5 16-bit operand lane layout: lane half h, element e ->
//   K = (e&7) + ((e>>3)<<4) + (h<<3)
__device__ __forceinline__ int kpat(int e, int half) {
  return (e & 7) + ((e >> 3) << 4) + (half << 3);
}

// ---------------------------------------------------------------------------
// Bilinear resize (NCHW planes flattened: P = B*C planes)
// ---------------------------------------------------------------------------
__global__ void resize_kernel(const float* __restrict__ in, float* __restrict__ out,
                              int P, int IH, int IW, int OH, int OW) {
  int idx = blockIdx.x * blockDim.x + threadIdx.x;
  int total = P * OH * OW;
  if (idx >= total) return;
  int ox = idx % OW;
  int t  = idx / OW;
  int oy = t % OH;
  int p  = t / OH;
  float sy = (oy + 0.5f) * ((float)IH / (float)OH) - 0.5f;
  float sx = (ox + 0.5f) * ((float)IW / (float)OW) - 0.5f;
  sy = fminf(fmaxf(sy, 0.0f), (float)(IH - 1));
  sx = fminf(fmaxf(sx, 0.0f), (float)(IW - 1));
  int y0 = (int)floorf(sy); y0 = min(max(y0, 0), IH - 1);
  int x0 = (int)floorf(sx); x0 = min(max(x0, 0), IW - 1);
  int y1 = min(y0 + 1, IH - 1);
  int x1 = min(x0 + 1, IW - 1);
  float wy = sy - (float)y0;
  float wx = sx - (float)x0;
  const float* pp = in + (size_t)p * IH * IW;
  float v00 = pp[y0 * IW + x0], v01 = pp[y0 * IW + x1];
  float v10 = pp[y1 * IW + x0], v11 = pp[y1 * IW + x1];
  float v0 = v00 * (1.0f - wx) + v01 * wx;
  float v1 = v10 * (1.0f - wx) + v11 * wx;
  out[idx] = v0 * (1.0f - wy) + v1 * wy;
}

// ---------------------------------------------------------------------------
// Pack OIHW fp32 weights -> f16 A-operand tiles in WMMA lane order.
// Apack[((ct*NCH + t)*32 + lane)*16 + e]
// ---------------------------------------------------------------------------
template<int Cin, int Cout, int Ksz>
__global__ void pack_a_kernel(const float* __restrict__ w, _Float16* __restrict__ Ap) {
  constexpr int kk = Ksz * Ksz;
  constexpr int Ktot = Cin * kk;
  constexpr int NCH = (Ktot + 31) / 32;
  constexpr int coTiles = (Cout + 15) / 16;
  constexpr int total = coTiles * NCH * 512;
  int idx = blockIdx.x * blockDim.x + threadIdx.x;
  if (idx >= total) return;
  int e    = idx & 15;
  int lane = (idx >> 4) & 31;
  int cidx = idx >> 9;
  int t  = cidx % NCH;
  int ct = cidx / NCH;
  int half = lane >> 4, l = lane & 15;
  int k = t * 32 + kpat(e, half);
  int m = ct * 16 + l;
  float v = 0.0f;
  if (k < Ktot && m < Cout) {
    int ci = k / kk, r = k - ci * kk, ky = r / Ksz, kx = r - ky * Ksz;
    v = w[((m * Cin + ci) * Ksz + ky) * Ksz + kx];
  }
  Ap[idx] = (_Float16)v;
}

// ---------------------------------------------------------------------------
// Pack im2col of NCHW fp32 activations -> f16 B-operand tiles (WMMA lane order).
// Bpack[(((b*pixTiles + pt)*NCH + t)*32 + lane)*16 + e]
// ---------------------------------------------------------------------------
template<int Cin, int Hin, int Win, int Stride, int Ksz>
__global__ void pack_b_kernel(const float* __restrict__ in, _Float16* __restrict__ Bp) {
  constexpr int pad  = (Ksz - 1) / 2;
  constexpr int Hout = (Hin + 2 * pad - Ksz) / Stride + 1;
  constexpr int Wout = (Win + 2 * pad - Ksz) / Stride + 1;
  constexpr int Npix = Hout * Wout;
  constexpr int kk   = Ksz * Ksz;
  constexpr int Ktot = Cin * kk;
  constexpr int NCH  = (Ktot + 31) / 32;
  constexpr int pixTiles = Npix / 16;
  constexpr int total = 2 * pixTiles * NCH * 512;
  int idx = blockIdx.x * blockDim.x + threadIdx.x;
  if (idx >= total) return;
  int e    = idx & 15;
  int lane = (idx >> 4) & 31;
  int cidx = idx >> 9;
  int t    = cidx % NCH;
  int rest = cidx / NCH;
  int pt   = rest % pixTiles;
  int b    = rest / pixTiles;
  int half = lane >> 4, l = lane & 15;
  int k = t * 32 + kpat(e, half);
  int n = pt * 16 + l;
  float v = 0.0f;
  if (k < Ktot) {
    int ci = k / kk, r = k - ci * kk, ky = r / Ksz, kx = r - ky * Ksz;
    int oy = n / Wout, ox = n % Wout;
    int iy = oy * Stride + ky - pad;
    int ix = ox * Stride + kx - pad;
    if (iy >= 0 && iy < Hin && ix >= 0 && ix < Win)
      v = in[((size_t)(b * Cin + ci) * Hin + iy) * Win + ix];
  }
  Bp[idx] = (_Float16)v;
}

// ---------------------------------------------------------------------------
// Pure WMMA GEMM on pre-packed operands: per k-step just two v16h (b128-pair)
// loads + one v_wmma_f32_16x16x32_f16. One wave per 16x16 output tile.
// ---------------------------------------------------------------------------
template<int Cout, int Npix, int NCH, bool Relu, bool HasBias, bool HasAdd>
__global__ __launch_bounds__(32) void gemm_wmma(
    const _Float16* __restrict__ Ap, const _Float16* __restrict__ Bp,
    const float* __restrict__ bias, const float* __restrict__ addv,
    float* __restrict__ out)
{
  constexpr int pixTiles = Npix / 16;
  const int lane = threadIdx.x;       // 0..31, wave32
  const int half = lane >> 4;
  const int l    = lane & 15;
  const int pt = blockIdx.x;
  const int ct = blockIdx.y;
  const int b  = blockIdx.z;

  const _Float16* ap = Ap + ((size_t)(ct * NCH) * 32 + lane) * 16;
  const _Float16* bp = Bp + (((size_t)(b * pixTiles + pt) * NCH) * 32 + lane) * 16;

  v8f acc = {0.f, 0.f, 0.f, 0.f, 0.f, 0.f, 0.f, 0.f};
#pragma unroll
  for (int t = 0; t < NCH; ++t) {
    v16h a  = *(const v16h*)(ap + (size_t)t * 512);
    v16h bm = *(const v16h*)(bp + (size_t)t * 512);
    acc = __builtin_amdgcn_wmma_f32_16x16x32_f16(false, a, false, bm,
                                                 (short)0, acc, false, false);
  }

  float add = 0.0f;
  if constexpr (HasAdd) add = addv[0];
  const int n = pt * 16 + l;
#pragma unroll
  for (int r = 0; r < 8; ++r) {
    const int mo = ct * 16 + r + 8 * half;
    if ((Cout % 16 == 0) || (mo < Cout)) {
      float v = acc[r];
      if constexpr (HasBias) v += bias[mo];
      if constexpr (Relu)    v  = fmaxf(v, 0.0f);
      if constexpr (HasAdd)  v += add;   // splat = relu(conv)+val
      out[(size_t)(b * Cout + mo) * Npix + n] = v;
    }
  }
}

// one conv layer = pack A + pack B + WMMA GEMM
template<int Cin, int Cout, int Hin, int Win, int Stride, int Ksz,
         bool Relu, bool HasBias, bool HasAdd>
static void conv_layer(const float* in, const float* w, const float* bias,
                       const float* addv, float* out,
                       _Float16* Apack, _Float16* Bpack, hipStream_t stream) {
  constexpr int pad  = (Ksz - 1) / 2;
  constexpr int Hout = (Hin + 2 * pad - Ksz) / Stride + 1;
  constexpr int Wout = (Win + 2 * pad - Ksz) / Stride + 1;
  constexpr int Npix = Hout * Wout;
  constexpr int Ktot = Cin * Ksz * Ksz;
  constexpr int NCH  = (Ktot + 31) / 32;
  constexpr int coTiles  = (Cout + 15) / 16;
  constexpr int pixTiles = Npix / 16;
  constexpr int totA = coTiles * NCH * 512;
  constexpr int totB = 2 * pixTiles * NCH * 512;
  pack_a_kernel<Cin, Cout, Ksz><<<(totA + 255) / 256, 256, 0, stream>>>(w, Apack);
  pack_b_kernel<Cin, Hin, Win, Stride, Ksz><<<(totB + 255) / 256, 256, 0, stream>>>(in, Bpack);
  gemm_wmma<Cout, Npix, NCH, Relu, HasBias, HasAdd>
      <<<dim3(pixTiles, coTiles, 2), 32, 0, stream>>>(Apack, Bpack, bias, addv, out);
}

// ---------------------------------------------------------------------------
// Fully connected:  out[b][o] = act( bias[o] + sum_k x[b][k]*w[o][k] )
// ---------------------------------------------------------------------------
__global__ void fc_kernel(const float* __restrict__ x, const float* __restrict__ w,
                          const float* __restrict__ bias, float* __restrict__ out,
                          int In, int Out, int doRelu) {
  int idx = blockIdx.x * blockDim.x + threadIdx.x;
  if (idx >= 2 * Out) return;
  int b = idx / Out, o = idx % Out;
  const float* xr = x + b * In;
  const float* wr = w + o * In;
  float s = bias[o];
  for (int k = 0; k < In; ++k) s += xr[k] * wr[k];
  out[idx] = doRelu ? fmaxf(s, 0.0f) : s;
}

// ---------------------------------------------------------------------------
// fusion = relu(gf[b][c] + loc[b][c][y][x])
// ---------------------------------------------------------------------------
__global__ void fusion_kernel(const float* __restrict__ loc, const float* __restrict__ gf,
                              float* __restrict__ out, int HW, int total) {
  int idx = blockIdx.x * blockDim.x + threadIdx.x;
  if (idx >= total) return;
  int bc = idx / HW;
  out[idx] = fmaxf(gf[bc] + loc[idx], 0.0f);
}

// ---------------------------------------------------------------------------
// Fused guide + bilateral slice + affine apply, 4 pixels/thread (float4 I/O).
// grid[b][c][z][y][x] == coeff[b][z*12+c][y][x]  (transpose folded into index)
// ---------------------------------------------------------------------------
__global__ void guide_slice_kernel(const float* __restrict__ image,
                                   const float* __restrict__ coeff,
                                   const float* __restrict__ ccm_w,
                                   const float* __restrict__ ccm_b,
                                   const float* __restrict__ shifts,
                                   const float* __restrict__ slopes,
                                   const float* __restrict__ proj_w,
                                   const float* __restrict__ proj_b,
                                   float* __restrict__ out) {
  const int H = 1024, W = 1024, GH = 16, GW = 16, L = 8, NC = 12;
  int t = blockIdx.x * blockDim.x + threadIdx.x;
  int xq   = t % (W / 4);
  int rest = t / (W / 4);
  int y = rest % H;
  int b = rest / H;
  if (b >= 2) return;
  int x0 = xq * 4;

  const size_t plane = (size_t)H * W;
  const float* base = image + (size_t)b * 3 * plane + (size_t)y * W + x0;
  // prefetch next row of each plane (global_prefetch_b8 path)
  if (y + 1 < H) {
    __builtin_prefetch(base + W, 0, 1);
    __builtin_prefetch(base + plane + W, 0, 1);
    __builtin_prefetch(base + 2 * plane + W, 0, 1);
  }
  float4 r4 = *(const float4*)(base);
  float4 g4 = *(const float4*)(base + plane);
  float4 b4 = *(const float4*)(base + 2 * plane);
  float rgbx[3][4] = {{r4.x, r4.y, r4.z, r4.w},
                      {g4.x, g4.y, g4.z, g4.w},
                      {b4.x, b4.y, b4.z, b4.w}};

  const float gy = ((float)y + 0.5f) * ((float)GH / (float)H) - 0.5f;
  const float fyf = floorf(gy);
  const int   fy  = (int)fyf;
  const float wy  = gy - fyf;

  float oc[3][4];

#pragma unroll
  for (int px = 0; px < 4; ++px) {
    const int x = x0 + px;
    // ----- guide: ccm -> piecewise-linear curve -> projection -> clip -----
    float g = proj_b[0];
#pragma unroll
    for (int c = 0; c < 3; ++c) {
      float gm = ccm_b[c];
#pragma unroll
      for (int i = 0; i < 3; ++i) gm += ccm_w[c * 3 + i] * rgbx[i][px];
      float cv = 0.0f;
      for (int p = 0; p < 16; ++p)
        cv += slopes[c * 16 + p] * fmaxf(gm - shifts[c * 16 + p], 0.0f);
      g += proj_w[c] * cv;
    }
    const float guide = fminf(fmaxf(g, 0.0f), 1.0f);

    // ----- trilinear slice of 12-channel grid -----
    const float gx  = ((float)x + 0.5f) * ((float)GW / (float)W) - 0.5f;
    const float gz  = guide * (float)L - 0.5f;
    const float fxf = floorf(gx), fzf = floorf(gz);
    const int   fx  = (int)fxf,  fz  = (int)fzf;
    const float wx  = gx - fxf,  wz  = gz - fzf;

    float acc[NC];
#pragma unroll
    for (int c = 0; c < NC; ++c) acc[c] = 0.0f;

#pragma unroll
    for (int dz = 0; dz < 2; ++dz) {
      int iz = min(max(fz + dz, 0), L - 1);
      float wz_ = dz ? wz : (1.0f - wz);
#pragma unroll
      for (int dy = 0; dy < 2; ++dy) {
        int iy = min(max(fy + dy, 0), GH - 1);
        float wzy = wz_ * (dy ? wy : (1.0f - wy));
#pragma unroll
        for (int dx = 0; dx < 2; ++dx) {
          int ix = min(max(fx + dx, 0), GW - 1);
          float wgt = wzy * (dx ? wx : (1.0f - wx));
          const float* cp = coeff + (((size_t)b * 96 + iz * NC) * GH + iy) * GW + ix;
#pragma unroll
          for (int c = 0; c < NC; ++c) acc[c] += cp[(size_t)c * GH * GW] * wgt;
        }
      }
    }

    // ----- per-pixel 3x4 affine apply + clip -----
#pragma unroll
    for (int c = 0; c < 3; ++c) {
      float v = acc[c * 4 + 3];
#pragma unroll
      for (int i = 0; i < 3; ++i) v += acc[c * 4 + i] * rgbx[i][px];
      oc[c][px] = fminf(fmaxf(v, 0.0f), 1.0f);
    }
  }

  float* ob = out + (size_t)b * 3 * plane + (size_t)y * W + x0;
#pragma unroll
  for (int c = 0; c < 3; ++c) {
    float4 v = {oc[c][0], oc[c][1], oc[c][2], oc[c][3]};
    *(float4*)(ob + (size_t)c * plane) = v;
  }
}

// ---------------------------------------------------------------------------
extern "C" void kernel_launch(void* const* d_in, const int* in_sizes, int n_in,
                              void* d_out, int out_size, void* d_ws, size_t ws_size,
                              hipStream_t stream) {
  (void)in_sizes; (void)n_in; (void)out_size; (void)ws_size;
  const float* image  = (const float*)d_in[0];
  const float* val    = (const float*)d_in[1];
  const float* s0_w   = (const float*)d_in[2];  const float* s0_b = (const float*)d_in[3];
  const float* s1_w   = (const float*)d_in[4];  const float* s1_b = (const float*)d_in[5];
  const float* s2_w   = (const float*)d_in[6];  const float* s2_b = (const float*)d_in[7];
  const float* s3_w   = (const float*)d_in[8];  const float* s3_b = (const float*)d_in[9];
  const float* g0_w   = (const float*)d_in[10]; const float* g0_b = (const float*)d_in[11];
  const float* g1_w   = (const float*)d_in[12]; const float* g1_b = (const float*)d_in[13];
  const float* fc0_w  = (const float*)d_in[14]; const float* fc0_b = (const float*)d_in[15];
  const float* fc1_w  = (const float*)d_in[16]; const float* fc1_b = (const float*)d_in[17];
  const float* l0_w   = (const float*)d_in[18]; const float* l0_b = (const float*)d_in[19];
  const float* l1_w   = (const float*)d_in[20];
  const float* pred_w = (const float*)d_in[21]; const float* pred_b = (const float*)d_in[22];
  const float* ccm_w  = (const float*)d_in[23]; const float* ccm_b  = (const float*)d_in[24];
  const float* shifts = (const float*)d_in[25]; const float* slopes = (const float*)d_in[26];
  const float* proj_w = (const float*)d_in[27]; const float* proj_b = (const float*)d_in[28];
  float* outp = (float*)d_out;

  float* ws = (float*)d_ws;
  float* lr     = ws;                // 2*3*256*256 = 393216
  float* s0o    = lr     + 393216;   // 2*8*128*128 = 262144
  float* s1o    = s0o    + 262144;   // 2*16*64*64  = 131072
  float* s2o    = s1o    + 131072;   // 2*32*32*32  = 65536
  float* splat  = s2o    + 65536;    // 2*64*16*16  = 32768
  float* g0o    = splat  + 32768;    // 2*16*8*8    = 2048
  float* g1o    = g0o    + 2048;     // 2*16*4*4    = 512
  float* fc0o   = g1o    + 512;      // 2*256       = 512
  float* gf     = fc0o   + 512;      // 2*64        = 128
  float* loc0   = gf     + 128;      // 32768
  float* loc1   = loc0   + 32768;    // 32768
  float* fus    = loc1   + 32768;    // 32768
  float* coeff  = fus    + 32768;    // 2*96*16*16  = 49152
  // f16 pack scratch (32B aligned: float offsets are multiples of 8)
  _Float16* Apack = (_Float16*)(coeff + 49152);          // <= 36864 f16
  _Float16* Bpack = (_Float16*)(coeff + 49152 + 18432);  // <= 1048576 f16

  // 1) low-res image
  {
    int total = 2 * 3 * 256 * 256;
    resize_kernel<<<(total + 255) / 256, 256, 0, stream>>>(image, lr, 6, 1024, 1024, 256, 256);
  }
  // 2) splat conv stack (pack + WMMA GEMM per layer)
  conv_layer<3,  8,  256, 256, 2, 3, true,  true,  false>(lr,    s0_w, s0_b, nullptr, s0o,  Apack, Bpack, stream);
  conv_layer<8,  16, 128, 128, 2, 3, true,  true,  false>(s0o,   s1_w, s1_b, nullptr, s1o,  Apack, Bpack, stream);
  conv_layer<16, 32, 64,  64,  2, 3, true,  true,  false>(s1o,   s2_w, s2_b, nullptr, s2o,  Apack, Bpack, stream);
  conv_layer<32, 64, 32,  32,  2, 3, true,  true,  true >(s2o,   s3_w, s3_b, val,     splat, Apack, Bpack, stream); // +val
  // 3) global path
  conv_layer<64, 16, 16,  16,  2, 3, true,  true,  false>(splat, g0_w, g0_b, nullptr, g0o,  Apack, Bpack, stream);
  conv_layer<16, 16, 8,   8,   2, 3, true,  true,  false>(g0o,   g1_w, g1_b, nullptr, g1o,  Apack, Bpack, stream);
  fc_kernel<<<2, 256, 0, stream>>>(g1o, fc0_w, fc0_b, fc0o, 256, 256, 1);
  fc_kernel<<<1, 128, 0, stream>>>(fc0o, fc1_w, fc1_b, gf, 256, 64, 0);
  // 4) local path
  conv_layer<64, 64, 16,  16,  1, 3, true,  true,  false>(splat, l0_w, l0_b, nullptr, loc0, Apack, Bpack, stream);
  conv_layer<64, 64, 16,  16,  1, 3, false, false, false>(loc0,  l1_w, nullptr, nullptr, loc1, Apack, Bpack, stream);
  // 5) fusion + prediction (coeff grid)
  fusion_kernel<<<(32768 + 255) / 256, 256, 0, stream>>>(loc1, gf, fus, 256, 32768);
  conv_layer<64, 96, 16,  16,  1, 1, false, true,  false>(fus,   pred_w, pred_b, nullptr, coeff, Apack, Bpack, stream);
  // 6) full-res fused guide + slice + apply (bandwidth stage)
  {
    int threads = 2 * 1024 * (1024 / 4);
    guide_slice_kernel<<<threads / 256, 256, 0, stream>>>(
        image, coeff, ccm_w, ccm_b, shifts, slopes, proj_w, proj_b, outp);
  }
}
